// EquiformerQM9PointCloud_87419764342918
// MI455X (gfx1250) — matrix operand order repaired and checked by reading, not compiled, not run on hardware
//
#include <hip/hip_runtime.h>

// ---------------------------------------------------------------------------
// Equiformer-QM9 forward, MI455X (gfx1250, wave32, WMMA).
// All C=128-deep projections run on v_wmma_f32_16x16x32_f16 (f16 A/B, f32 acc).
// GEMM: 128x64 tile per 8-wave block; weight tile async-staged to LDS
// (global_load_async_to_lds_b128 + s_wait_asynccnt), B fragments from LDS,
// 16x64 per wave (A fragment reused across 4 accumulators). Per K-step all
// four B fragments are loaded before the WMMA chain to overlap DS latency.
// Working set ~70MB -> L2 resident; compute path is the WMMA ceiling.
// ---------------------------------------------------------------------------

#define BB   16
#define NM   1024
#define HH   4
#define DHD  32
#define KNN  16
#define CC   128
#define TT   8192
#define ROWS (BB * NM)      // 16384 dense rows
#define QKVW (4 * CC)       // 512 fused QKV output width

typedef __attribute__((ext_vector_type(16))) _Float16 v16h;
typedef __attribute__((ext_vector_type(8)))  _Float16 v8h;
typedef __attribute__((ext_vector_type(8)))  float    v8f;

// ---------------- workspace layout (bytes) ----------------
static constexpr size_t OFF_F0D   = 0;                                   // f32 [ROWS][CC]
static constexpr size_t OFF_F0H   = OFF_F0D   + (size_t)ROWS * CC * 4;   // f16 [ROWS][CC]
static constexpr size_t OFF_COORD = OFF_F0H   + (size_t)ROWS * CC * 2;   // f32 [ROWS][3]
static constexpr size_t OFF_CNT   = OFF_COORD + (size_t)ROWS * 3 * 4;    // i32 [16]
static constexpr size_t OFF_START = OFF_CNT   + 64;                      // i32 [16]
static constexpr size_t OFF_NBR   = OFF_START + 64;                      // i32 [ROWS][KNN]
static constexpr size_t OFF_DIST  = OFF_NBR   + (size_t)ROWS * KNN * 4;  // f32 [ROWS][KNN]
static constexpr size_t OFF_DIREC = OFF_DIST  + (size_t)ROWS * KNN * 4;  // f32 [ROWS][KNN][3]
static constexpr size_t OFF_WQKVT = OFF_DIREC + (size_t)ROWS * KNN * 12; // f16 [512][128] (transposed)
static constexpr size_t OFF_WO0T  = OFF_WQKVT + (size_t)QKVW * CC * 2;   // f16 [128][128]
static constexpr size_t OFF_WO1T  = OFF_WO0T  + (size_t)CC * CC * 2;
static constexpr size_t OFF_WGT   = OFF_WO1T  + (size_t)CC * CC * 2;
static constexpr size_t OFF_QKV   = OFF_WGT   + (size_t)CC * CC * 2;     // f32 [ROWS][512]
static constexpr size_t OFF_OUT0H = OFF_QKV   + (size_t)ROWS * QKVW * 4; // f16 [ROWS][CC]
static constexpr size_t OFF_OUT1H = OFF_OUT0H + (size_t)ROWS * CC * 2;   // f16 3x[ROWS][CC]
static constexpr size_t OFF_INVH  = OFF_OUT1H + 3 * (size_t)ROWS * CC * 2;
// aliases into the (dead after attention) QKV region:
static constexpr size_t OFF_O0    = OFF_QKV;                             // f32 [ROWS][CC]
static constexpr size_t OFF_F1X   = OFF_QKV + 1 * (size_t)ROWS * CC * 4;
static constexpr size_t OFF_F1Y   = OFF_QKV + 2 * (size_t)ROWS * CC * 4;
static constexpr size_t OFF_F1Z   = OFF_QKV + 3 * (size_t)ROWS * CC * 4;

// ---------------- gfx1250 async-LDS helpers --------------------------------
static __device__ __forceinline__ void eq_wait_async0() {
#if __has_builtin(__builtin_amdgcn_s_wait_asynccnt)
  __builtin_amdgcn_s_wait_asynccnt(0);
#else
  asm volatile("s_wait_asynccnt 0x0" ::: "memory");
#endif
}

// one lane moves 16B global -> LDS, tracked by ASYNCcnt
static __device__ __forceinline__ void eq_async_g2l_b128(unsigned lds_off, const void* gaddr) {
  asm volatile("global_load_async_to_lds_b128 %0, %1, off"
               :: "v"(lds_off), "v"(gaddr)
               : "memory");
}

// ---------------- segment info: counts + starts (batch is sorted) ----------
__global__ void eq_seg_kernel(const int* __restrict__ batch,
                              int* __restrict__ counts, int* __restrict__ starts) {
  int t = blockIdx.x * blockDim.x + threadIdx.x;
  if (t >= TT) return;
  int b = batch[t];
  atomicAdd(&counts[b], 1);
  atomicMin(&starts[b], t);
}

// ---------------- embed + dense-batch scatter ------------------------------
__global__ void eq_embed_kernel(const float* __restrict__ x, const float* __restrict__ pos,
                                const int* __restrict__ batch, const int* __restrict__ starts,
                                const float* __restrict__ embW, const float* __restrict__ embB,
                                float* __restrict__ f0d, _Float16* __restrict__ f0h,
                                float* __restrict__ coords) {
  int id = blockIdx.x * blockDim.x + threadIdx.x;  // over TT*CC
  if (id >= TT * CC) return;
  int t = id >> 7, c = id & (CC - 1);
  int b = batch[t];
  int n = t - starts[b];
  size_t row = (size_t)b * NM + n;
  float v = embB[c];
  #pragma unroll
  for (int j = 0; j < 11; ++j) v += x[t * 11 + j] * embW[j * CC + c];
  f0d[row * CC + c] = v;
  f0h[row * CC + c] = (_Float16)v;
  if (c < 3) coords[row * 3 + c] = pos[t * 3 + c];
}

// ---------------- convert weights to f16, transposed ([out][in]) -----------
__global__ void eq_wcvt_kernel(const float* __restrict__ Wq, const float* __restrict__ Wk,
                               const float* __restrict__ Wv0, const float* __restrict__ Wv1,
                               const float* __restrict__ Wo0, const float* __restrict__ Wo1,
                               const float* __restrict__ Wg,
                               _Float16* __restrict__ qkvt, _Float16* __restrict__ wo0t,
                               _Float16* __restrict__ wo1t, _Float16* __restrict__ wgt) {
  int id = blockIdx.x * blockDim.x + threadIdx.x;
  int NQKV = QKVW * CC;         // 65536
  int NSQ  = CC * CC;           // 16384
  if (id < NQKV) {
    int n = id >> 7, k = id & (CC - 1);
    int sel = n >> 7, col = n & (CC - 1);
    const float* W = (sel == 0) ? Wq : (sel == 1) ? Wk : (sel == 2) ? Wv0 : Wv1;
    qkvt[(size_t)n * CC + k] = (_Float16)W[k * CC + col];
    return;
  }
  id -= NQKV;
  if (id < NSQ) { int n = id >> 7, k = id & (CC-1); wo0t[(size_t)n*CC+k] = (_Float16)Wo0[k*CC+n]; return; }
  id -= NSQ;
  if (id < NSQ) { int n = id >> 7, k = id & (CC-1); wo1t[(size_t)n*CC+k] = (_Float16)Wo1[k*CC+n]; return; }
  id -= NSQ;
  if (id < NSQ) { int n = id >> 7, k = id & (CC-1); wgt [(size_t)n*CC+k] = (_Float16)Wg [k*CC+n]; }
}

// ---------------- brute-force kNN (register insertion network) -------------
__global__ void eq_knn_kernel(const float* __restrict__ coords, const int* __restrict__ counts,
                              int* __restrict__ nbr, float* __restrict__ dist,
                              float* __restrict__ direc) {
  int id = blockIdx.x * blockDim.x + threadIdx.x;  // over ROWS
  if (id >= ROWS) return;
  int b = id >> 10, n = id & (NM - 1);
  int cnt = counts[b];
  if (n >= cnt) return;
  size_t row = (size_t)id;
  float qx = coords[row * 3 + 0], qy = coords[row * 3 + 1], qz = coords[row * 3 + 2];
  float bd[KNN]; int bi[KNN];
  #pragma unroll
  for (int i = 0; i < KNN; ++i) { bd[i] = 3e38f; bi[i] = -1; }
  size_t base = (size_t)b * NM;
  for (int m = 0; m < cnt; ++m) {
    float dx = qx - coords[(base + m) * 3 + 0];
    float dy = qy - coords[(base + m) * 3 + 1];
    float dz = qz - coords[(base + m) * 3 + 2];
    float cd = dx * dx + dy * dy + dz * dz;
    int ci = m;
    #pragma unroll
    for (int s = 0; s < KNN; ++s) {  // sorted insertion, evicted max falls out
      if (cd < bd[s]) { float td = bd[s]; bd[s] = cd; cd = td; int ti = bi[s]; bi[s] = ci; ci = ti; }
    }
  }
  #pragma unroll
  for (int k = 0; k < KNN; ++k) {
    int m = bi[k];
    nbr[row * KNN + k] = m;
    if (m >= 0) {
      float d = sqrtf(bd[k] + 1e-6f);
      float rx = qx - coords[(base + m) * 3 + 0];
      float ry = qy - coords[(base + m) * 3 + 1];
      float rz = qz - coords[(base + m) * 3 + 2];
      dist[row * KNN + k] = d;
      float inv = 1.0f / d;
      direc[(row * KNN + k) * 3 + 0] = rx * inv;
      direc[(row * KNN + k) * 3 + 1] = ry * inv;
      direc[(row * KNN + k) * 3 + 2] = rz * inv;
    } else {
      dist[row * KNN + k] = 0.f;
      direc[(row * KNN + k) * 3 + 0] = 0.f;
      direc[(row * KNN + k) * 3 + 1] = 0.f;
      direc[(row * KNN + k) * 3 + 2] = 0.f;
    }
  }
}

// ---------------- WMMA GEMM: C(f32[M][N]) = A(f16[M][K=128]) x Bt(f16[N][K]) -
// Block: 8 waves -> 128 rows x 64 cols. The 64x128 f16 weight tile is
// async-copied to LDS once (ASYNCcnt), B fragments read via ds_load_b128
// with a padded pitch (136 halves) so the 16 fragment lanes hit 16 banks.
// Each wave: 16x64 tile = 4 f32 accumulators, A fragment reused 4x.
#define GM_WAVES 8
#define BPITCH   (CC + 8)     // halves; 272B row pitch in LDS
__global__ void eq_wmma_gemm(const _Float16* __restrict__ A, const _Float16* __restrict__ Bt,
                             const float* __restrict__ Cin, float* __restrict__ Cout,
                             int M, int N, int Kd /* must be 128 */) {
  __shared__ _Float16 bsh[64 * BPITCH];               // 17 KB
  const int tid  = threadIdx.x;
  const int lane = tid & 31;
  const int wave = tid >> 5;
  const int rowBase = blockIdx.x * (16 * GM_WAVES);   // 128 rows per block
  const int colBase = blockIdx.y * 64;                // 64 cols per block
  const int g = lane >> 4;
  const int l = lane & 15;

  // ---- stage weight tile: 64 rows x 256B, 1024 16B-chunks, 4 per thread ----
  {
    const char* gsrc = (const char*)(Bt + (size_t)colBase * CC);
    unsigned lbase = (unsigned)(uintptr_t)(&bsh[0]);  // flat->LDS: low 32 bits
    #pragma unroll
    for (int it = 0; it < 4; ++it) {
      int ch = it * 256 + tid;
      int rr = ch >> 4;           // weight row   (0..63)
      int cw = ch & 15;           // 16B chunk in row
      unsigned loff = lbase + (unsigned)(rr * (BPITCH * 2) + cw * 16);
      eq_async_g2l_b128(loff, gsrc + rr * (CC * 2) + cw * 16);
    }
    eq_wait_async0();
    __syncthreads();
  }

  const int tm = rowBase + wave * 16;                 // this wave's 16-row stripe
  v8f acc[4];
  #pragma unroll
  for (int nt = 0; nt < 4; ++nt) {
    #pragma unroll
    for (int r = 0; r < 8; ++r) {
      int m = tm + r + 8 * g;                         // C/D layout: m = r + 8g
      acc[nt][r] = Cin ? Cin[(size_t)m * N + colBase + nt * 16 + l] : 0.0f;
    }
  }

  #pragma unroll
  for (int kk = 0; kk < CC; kk += 32) {
    // A fragment: lane row = tm+l; halves 0..7 -> K=kk+8g.., 8..15 -> K=kk+16+8g..
    const _Float16* pa = A + (size_t)(tm + l) * Kd + kk + 8 * g;
    v8h a0 = *(const v8h*)pa;
    v8h a1 = *(const v8h*)(pa + 16);
    v16h af;
    #pragma unroll
    for (int i = 0; i < 8; ++i) { af[i] = a0[i]; af[i + 8] = a1[i]; }

    // Load ALL four B fragments first (8x ds_load_b128 in flight), then the
    // 4-WMMA chain: overlaps LDS latency with matrix issue instead of a
    // dscnt==0 stall in front of every WMMA.
    v8h b0[4], b1[4];
    #pragma unroll
    for (int nt = 0; nt < 4; ++nt) {
      const _Float16* pb = &bsh[(nt * 16 + l) * BPITCH + kk + 8 * g];
      b0[nt] = *(const v8h*)pb;
      b1[nt] = *(const v8h*)(pb + 16);
    }
    #pragma unroll
    for (int nt = 0; nt < 4; ++nt) {
      v16h bf;
      #pragma unroll
      for (int i = 0; i < 8; ++i) { bf[i] = b0[nt][i]; bf[i + 8] = b1[nt][i]; }
      acc[nt] = __builtin_amdgcn_wmma_f32_16x16x32_f16(false, af, false, bf, (short)0,
                                                       acc[nt], false, false);
    }
  }

  #pragma unroll
  for (int nt = 0; nt < 4; ++nt) {
    #pragma unroll
    for (int r = 0; r < 8; ++r) {
      int m = tm + r + 8 * g;
      Cout[(size_t)m * N + colBase + nt * 16 + l] = acc[nt][r];
    }
  }
}

// ---------------- kNN attention: softmax over K=16, deg-0 and deg-1 outputs -
__global__ void eq_attn_kernel(const float* __restrict__ qkv, const int* __restrict__ counts,
                               const int* __restrict__ nbr, const float* __restrict__ dist,
                               const float* __restrict__ direc, const float* __restrict__ dist_scale,
                               _Float16* __restrict__ out0h, _Float16* __restrict__ out1h) {
  int id = blockIdx.x * blockDim.x + threadIdx.x;  // over ROWS*HH
  if (id >= ROWS * HH) return;
  int h = id & (HH - 1);
  int row = id >> 2;
  int b = row >> 10, n = row & (NM - 1);
  int cnt = counts[b];
  _Float16* o0p = out0h + (size_t)row * CC + h * DHD;
  _Float16* oxp = out1h + 0 * (size_t)ROWS * CC + (size_t)row * CC + h * DHD;
  _Float16* oyp = out1h + 1 * (size_t)ROWS * CC + (size_t)row * CC + h * DHD;
  _Float16* ozp = out1h + 2 * (size_t)ROWS * CC + (size_t)row * CC + h * DHD;
  if (n >= cnt) {
    #pragma unroll
    for (int d = 0; d < DHD; ++d) { o0p[d] = (_Float16)0.f; oxp[d] = (_Float16)0.f; oyp[d] = (_Float16)0.f; ozp[d] = (_Float16)0.f; }
    return;
  }
  float q[DHD];
  const float* qp = qkv + (size_t)row * QKVW + h * DHD;
  #pragma unroll
  for (int d = 0; d < DHD; ++d) q[d] = qp[d];
  float sp = log1pf(expf(dist_scale[h]));
  const int* nb = nbr + (size_t)row * KNN;
  const float* dv = dist + (size_t)row * KNN;
  float lg[KNN];
  #pragma unroll
  for (int k = 0; k < KNN; ++k) {
    int m = nb[k];
    if (m >= 0) {
      const float* kp = qkv + (size_t)((size_t)b * NM + m) * QKVW + CC + h * DHD;
      float s = 0.f;
      #pragma unroll
      for (int d = 0; d < DHD; ++d) s += q[d] * kp[d];
      lg[k] = s * 0.17677669529663689f - sp * dv[k];
    } else {
      lg[k] = -1e9f;
    }
  }
  float mx = -3e38f;
  #pragma unroll
  for (int k = 0; k < KNN; ++k) mx = fmaxf(mx, lg[k]);
  float den = 0.f;
  #pragma unroll
  for (int k = 0; k < KNN; ++k) { lg[k] = expf(lg[k] - mx); den += lg[k]; }
  float rden = 1.0f / den;
  float o0[DHD], ox[DHD], oy[DHD], oz[DHD];
  #pragma unroll
  for (int d = 0; d < DHD; ++d) { o0[d] = 0.f; ox[d] = 0.f; oy[d] = 0.f; oz[d] = 0.f; }
  #pragma unroll
  for (int k = 0; k < KNN; ++k) {
    int m = nb[k];
    if (m < 0) continue;
    float a = lg[k] * rden;
    const float* v0p = qkv + (size_t)((size_t)b * NM + m) * QKVW + 2 * CC + h * DHD;
    const float* v1p = qkv + (size_t)((size_t)b * NM + m) * QKVW + 3 * CC + h * DHD;
    const float* dp = direc + ((size_t)row * KNN + k) * 3;
    float dx = dp[0], dy = dp[1], dz = dp[2];
    #pragma unroll
    for (int d = 0; d < DHD; ++d) {
      o0[d] += a * v0p[d];
      float w = a * v1p[d];
      ox[d] += w * dx; oy[d] += w * dy; oz[d] += w * dz;
    }
  }
  #pragma unroll
  for (int d = 0; d < DHD; ++d) {
    o0p[d] = (_Float16)o0[d]; oxp[d] = (_Float16)ox[d];
    oyp[d] = (_Float16)oy[d]; ozp[d] = (_Float16)oz[d];
  }
}

// ---------------- invariant norm + residual base ----------------------------
__global__ void eq_inv_kernel(const float* __restrict__ f1x, const float* __restrict__ f1y,
                              const float* __restrict__ f1z, const float* __restrict__ o0,
                              float* __restrict__ f0d, _Float16* __restrict__ invh) {
  int id = blockIdx.x * blockDim.x + threadIdx.x;  // over ROWS*CC
  if (id >= ROWS * CC) return;
  float fx = f1x[id], fy = f1y[id], fz = f1z[id];
  invh[id] = (_Float16)sqrtf(fx * fx + fy * fy + fz * fz + 1e-6f);
  f0d[id] += o0[id];                                // base = f0 + out0@Wo0
}

// ---------------- masked mean pool + linear head ----------------------------
__global__ void eq_pool_kernel(const float* __restrict__ f0out, const int* __restrict__ counts,
                               const float* __restrict__ headW, const float* __restrict__ headB,
                               float* __restrict__ out) {
  __shared__ float pooled[CC];
  int b = blockIdx.x;
  int c = threadIdx.x;
  int cnt = counts[b];
  float s = 0.f;
  for (int n = 0; n < cnt; ++n) s += f0out[((size_t)b * NM + n) * CC + c];
  pooled[c] = s / fmaxf((float)cnt, 1.0f);
  __syncthreads();
  if (c < 19) {
    float o = headB[c];
    #pragma unroll 4
    for (int k = 0; k < CC; ++k) o += pooled[k] * headW[k * 19 + c];
    out[b * 19 + c] = o;
  }
}

// ---------------------------------------------------------------------------
extern "C" void kernel_launch(void* const* d_in, const int* in_sizes, int n_in,
                              void* d_out, int out_size, void* d_ws, size_t ws_size,
                              hipStream_t stream) {
  (void)in_sizes; (void)n_in; (void)out_size; (void)ws_size;
  const float* x       = (const float*)d_in[0];
  const float* pos     = (const float*)d_in[1];
  const int*   batch   = (const int*)  d_in[2];
  const float* embW    = (const float*)d_in[3];
  const float* embB    = (const float*)d_in[4];
  const float* Wq      = (const float*)d_in[5];
  const float* Wk      = (const float*)d_in[6];
  const float* Wv0     = (const float*)d_in[7];
  const float* Wv1     = (const float*)d_in[8];
  const float* Wo0     = (const float*)d_in[9];
  const float* Wo1     = (const float*)d_in[10];
  const float* Wg      = (const float*)d_in[11];
  const float* dscale  = (const float*)d_in[12];
  const float* headW   = (const float*)d_in[13];
  const float* headB   = (const float*)d_in[14];
  float* out = (float*)d_out;

  char* ws = (char*)d_ws;
  float*     f0d    = (float*)    (ws + OFF_F0D);
  _Float16*  f0h    = (_Float16*) (ws + OFF_F0H);
  float*     coords = (float*)    (ws + OFF_COORD);
  int*       counts = (int*)      (ws + OFF_CNT);
  int*       starts = (int*)      (ws + OFF_START);
  int*       nbr    = (int*)      (ws + OFF_NBR);
  float*     dist   = (float*)    (ws + OFF_DIST);
  float*     direc  = (float*)    (ws + OFF_DIREC);
  _Float16*  wqkvt  = (_Float16*) (ws + OFF_WQKVT);
  _Float16*  wo0t   = (_Float16*) (ws + OFF_WO0T);
  _Float16*  wo1t   = (_Float16*) (ws + OFF_WO1T);
  _Float16*  wgt    = (_Float16*) (ws + OFF_WGT);
  float*     qkv    = (float*)    (ws + OFF_QKV);
  _Float16*  out0h  = (_Float16*) (ws + OFF_OUT0H);
  _Float16*  out1h  = (_Float16*) (ws + OFF_OUT1H);
  _Float16*  invh   = (_Float16*) (ws + OFF_INVH);
  float*     o0     = (float*)    (ws + OFF_O0);   // aliases qkv (dead after attn)
  float*     f1x    = (float*)    (ws + OFF_F1X);
  float*     f1y    = (float*)    (ws + OFF_F1Y);
  float*     f1z    = (float*)    (ws + OFF_F1Z);

  // init
  hipMemsetAsync(counts, 0, 64, stream);
  hipMemsetAsync(starts, 0x7F, 64, stream);                 // ~INT_MAX per int
  hipMemsetAsync(f0d, 0, (size_t)ROWS * CC * 4, stream);
  hipMemsetAsync(f0h, 0, (size_t)ROWS * CC * 2, stream);

  // segment info
  eq_seg_kernel<<<(TT + 255) / 256, 256, 0, stream>>>(batch, counts, starts);

  // embed + scatter
  eq_embed_kernel<<<(TT * CC + 255) / 256, 256, 0, stream>>>(
      x, pos, batch, starts, embW, embB, f0d, f0h, coords);

  // weight convert/transpose
  {
    int total = QKVW * CC + 3 * CC * CC;
    eq_wcvt_kernel<<<(total + 255) / 256, 256, 0, stream>>>(
        Wq, Wk, Wv0, Wv1, Wo0, Wo1, Wg, wqkvt, wo0t, wo1t, wgt);
  }

  // kNN
  eq_knn_kernel<<<(ROWS + 255) / 256, 256, 0, stream>>>(coords, counts, nbr, dist, direc);

  // fused QKV projection: [16384 x 128] x [128 x 512]  (WMMA)
  {
    dim3 grid(ROWS / 128, QKVW / 64);
    eq_wmma_gemm<<<grid, 256, 0, stream>>>(f0h, wqkvt, nullptr, qkv, ROWS, QKVW, CC);
  }

  // attention
  eq_attn_kernel<<<(ROWS * HH + 255) / 256, 256, 0, stream>>>(
      qkv, counts, nbr, dist, direc, dscale, out0h, out1h);

  // out0 @ Wo0 and the three out1 planes @ Wo1 (WMMA, outputs alias qkv region)
  {
    dim3 grid(ROWS / 128, CC / 64);
    eq_wmma_gemm<<<grid, 256, 0, stream>>>(out0h, wo0t, nullptr, o0, ROWS, CC, CC);
    eq_wmma_gemm<<<grid, 256, 0, stream>>>(out1h + 0 * (size_t)ROWS * CC, wo1t, nullptr, f1x, ROWS, CC, CC);
    eq_wmma_gemm<<<grid, 256, 0, stream>>>(out1h + 1 * (size_t)ROWS * CC, wo1t, nullptr, f1y, ROWS, CC, CC);
    eq_wmma_gemm<<<grid, 256, 0, stream>>>(out1h + 2 * (size_t)ROWS * CC, wo1t, nullptr, f1z, ROWS, CC, CC);
  }

  // invariant norm -> f16, base = f0 + o0 (in-place into f0d)
  eq_inv_kernel<<<(ROWS * CC + 255) / 256, 256, 0, stream>>>(f1x, f1y, f1z, o0, f0d, invh);

  // gate: f0_out = base + inv @ Wg  (WMMA with C accumulate, in-place tiles)
  {
    dim3 grid(ROWS / 128, CC / 64);
    eq_wmma_gemm<<<grid, 256, 0, stream>>>(invh, wgt, f0d, f0d, ROWS, CC, CC);
  }

  // pool + head
  eq_pool_kernel<<<BB, CC, 0, stream>>>(f0d, counts, headW, headB, out);
}
